// PagedAttention_11699490914634
// MI455X (gfx1250) — matrix operand order repaired and checked by reading, not compiled, not run on hardware
//
#include <hip/hip_runtime.h>
#include <hip/hip_bf16.h>
#include <math.h>

// Problem constants (from reference)
#define B_   32
#define NIN  4096
#define H_   32
#define G_   8
#define HD_  128
#define BLOCKS_PER_SEQ_ 128
#define RHEADS 4            // H_/G_ query heads per KV group
#define QK_SCALE 0.08838834764831845f  // 1/sqrt(128)

typedef __attribute__((ext_vector_type(16))) __bf16        v16bf;
typedef __attribute__((ext_vector_type(8)))  float         v8f;
typedef __attribute__((ext_vector_type(4)))  unsigned int  v4u;
typedef __attribute__((ext_vector_type(8)))  int           v8i;
typedef __attribute__((ext_vector_type(4)))  int           v4i;

#if __has_builtin(__builtin_amdgcn_tensor_load_to_lds)
#define USE_TDM 1
#endif

// Convert 8 consecutive fp32 -> bf16 into dst[base..base+7]
__device__ __forceinline__ void cvt8(v16bf& d, int base, const float* __restrict__ p) {
  const float4* q = (const float4*)p;
  float4 f0 = q[0], f1 = q[1];
  d[base+0] = (__bf16)f0.x; d[base+1] = (__bf16)f0.y;
  d[base+2] = (__bf16)f0.z; d[base+3] = (__bf16)f0.w;
  d[base+4] = (__bf16)f1.x; d[base+5] = (__bf16)f1.y;
  d[base+6] = (__bf16)f1.z; d[base+7] = (__bf16)f1.w;
}
// Convert 16 consecutive fp32 -> full v16bf
__device__ __forceinline__ void cvt16(v16bf& d, const float* __restrict__ p) {
  cvt8(d, 0, p); cvt8(d, 8, p + 8);
}

// ---------------------------------------------------------------------------
// Generic Out[M=32,N] = X[32,K] * W[N,K]^T via v_wmma_f32_16x16x32_bf16.
// One wave computes one 16x16 output tile; K unrolled by 64 into TWO
// independent accumulator chains so consecutive WMMAs have no RAW hazard.
// fp32 -> bf16 conversion in registers; the fp32 weight stream stays the
// only HBM traffic (memory-bound by design).
// ---------------------------------------------------------------------------
__global__ __launch_bounds__(256) void gemm_xwT_bf16(
    const float* __restrict__ X, const float* __restrict__ W,
    float* __restrict__ Out, int N, int K)
{
  const int lane   = threadIdx.x & 31;
  const int waveId = blockIdx.x * 8 + (threadIdx.x >> 5);
  const int nTiles = N >> 4;
  if (waveId >= 2 * nTiles) return;
  const int tileN = waveId % nTiles;
  const int tileM = waveId / nTiles;   // 0 or 1 (M = 32)
  const int hi = lane >> 4;
  const int lo = lane & 15;

  const float* __restrict__ xrow = X + (size_t)(tileM * 16 + lo) * K;
  const float* __restrict__ wrow = W + (size_t)(tileN * 16 + lo) * K;

  v8f c0 = {}, c1 = {};
  for (int k0 = 0; k0 < K; k0 += 64) {
    __builtin_prefetch(wrow + k0 + 512, 0, 1);   // global_prefetch_b8, ~2KB ahead
    v16bf a0, b0, a1, b1;
    cvt16(a0,    xrow + k0 + hi * 16);
    cvt16(b0,    wrow + k0 + hi * 16);
    cvt16(a1,    xrow + k0 + 32 + hi * 16);
    cvt16(b1,    wrow + k0 + 32 + hi * 16);
    c0 = __builtin_amdgcn_wmma_f32_16x16x32_bf16(false, a0, false, b0,
                                                 (short)0, c0, false, false);
    c1 = __builtin_amdgcn_wmma_f32_16x16x32_bf16(false, a1, false, b1,
                                                 (short)0, c1, false, false);
  }
  v8f c = c0 + c1;
  float* __restrict__ ocol = Out + tileN * 16 + lo;
  #pragma unroll
  for (int i = 0; i < 8; ++i)
    ocol[(size_t)(tileM * 16 + hi * 8 + i) * N] = c[i];
}

// ---------------------------------------------------------------------------
// RoPE (NeoX-style halves) in place on q and k_new; folds 1/sqrt(HD) into q.
// grid: (B, H+G), 64 threads (pair d, d+64)
// ---------------------------------------------------------------------------
__global__ __launch_bounds__(64) void rope_scale(
    float* __restrict__ q, float* __restrict__ k,
    const float* __restrict__ cs, const float* __restrict__ sn)
{
  const int b = blockIdx.x;
  const int head = blockIdx.y;
  const int d = threadIdx.x;           // 0..63
  float* t; float sc;
  if (head < H_) { t = q + (size_t)(b * H_ + head) * HD_;        sc = QK_SCALE; }
  else           { t = k + (size_t)(b * G_ + (head - H_)) * HD_; sc = 1.0f;     }
  const float* cb = cs + b * HD_;
  const float* sb = sn + b * HD_;
  float t1 = t[d], t2 = t[d + 64];
  float o1 = (t1 * cb[d]      - t2 * sb[d])      * sc;
  float o2 = (t2 * cb[d + 64] + t1 * sb[d + 64]) * sc;
  t[d] = o1; t[d + 64] = o2;
}

// ---------------------------------------------------------------------------
// Paged attention, one workgroup per (b, g); 4 waves split the key range
// (flash-decoding) and combine through LDS.
//  - QK^T per 16-key tile via 4x v_wmma_f32_16x16x32_bf16 in two independent
//    accumulator chains
//  - V tile (16x128 fp32, uniform address) staged LDS-side by the Tensor
//    Data Mover, issued at the top of the loop and waited with
//    s_wait_tensorcnt just before the AV pass (overlaps with score WMMAs)
//  - online softmax with wave shfl reductions; probs broadcast via LDS
//  - the new token's K/V row (physical slot == slot_mapping[b]) is taken
//    from the freshly projected k_new/v_new (caches are read-only inputs)
// ---------------------------------------------------------------------------
__global__ __launch_bounds__(128) void paged_attn(
    const float* __restrict__ qv,   // B x H x HD (rope'd, pre-scaled)
    const float* __restrict__ knew, // B x G x HD (rope'd)
    const float* __restrict__ vnew, // B x G x HD
    const float* __restrict__ kc,   // NUM_BLOCKS x 16 x G x HD
    const float* __restrict__ vc,
    const int* __restrict__ btab,   // B x BLOCKS_PER_SEQ
    const int* __restrict__ ctxlen, // B
    const int* __restrict__ slotmap,// B
    float* __restrict__ o)          // B x H x HD
{
  __shared__ float4 lds_p[4][16];            // per-wave probs [key] -> (h0..h3)
  __shared__ float  lds_m[4][RHEADS];
  __shared__ float  lds_l[4][RHEADS];
  __shared__ float  lds_acc[4][RHEADS][HD_];
#ifdef USE_TDM
  __shared__ float  lds_v[4][16][HD_];       // 32 KB: one V tile per wave
#endif

  const int b = blockIdx.x, g = blockIdx.y;
  const int w    = threadIdx.x >> 5;
  const int lane = threadIdx.x & 31;
  const int hi = lane >> 4, lo = lane & 15;
  const int ctx  = ctxlen[b];
  const int slot = slotmap[b];
  const int nTiles = (ctx + 15) >> 4;

  // Q A-operands (16x32 bf16 per 32-dim chunk): rows lo<4 real, 4..15 zero.
  v16bf aq[4];
  const float* qrow = qv + (size_t)(b * H_ + g * RHEADS + lo) * HD_;
  #pragma unroll
  for (int cch = 0; cch < 4; ++cch) {
    if (lo < RHEADS) {
      cvt16(aq[cch], qrow + cch * 32 + hi * 16);
    } else {
      #pragma unroll
      for (int j = 0; j < 16; ++j) aq[cch][j] = (__bf16)0.f;
    }
  }

  float m_run[RHEADS], l_run[RHEADS], acc[RHEADS][4];
  #pragma unroll
  for (int h = 0; h < RHEADS; ++h) {
    m_run[h] = -1e30f; l_run[h] = 0.f;
    acc[h][0] = acc[h][1] = acc[h][2] = acc[h][3] = 0.f;
  }

  const float* knr = knew + (size_t)(b * G_ + g) * HD_;
  const float* vnr = vnew + (size_t)(b * G_ + g) * HD_;

  for (int st = w; st < nTiles; st += 4) {
    const int s0  = st << 4;                        // 16-key tile == one cache block
    const int blk = btab[b * BLOCKS_PER_SEQ_ + st];

#ifdef USE_TDM
    // Prior iteration's LDS V-reads must complete before TDM rewrites the tile.
    asm volatile("s_wait_dscnt 0" ::: "memory");
    {
      // D# for a 2D tile: 128 elems/row (fp32), 16 rows, row stride G*HD elems.
      const uint64_t gaddr = (uint64_t)(const void*)(vc + ((size_t)blk * 16 * G_ + g) * HD_);
      const uint32_t laddr = (uint32_t)(size_t)(&lds_v[w][0][0]);
      v4u g0 = { 1u,                                  // count=1, user descriptor
                 laddr,                               // lds_addr (bytes)
                 (unsigned int)gaddr,                 // global_addr[31:0]
                 (unsigned int)(gaddr >> 32) | 0x80000000u }; // addr[56:32] | type=2
      v8i g1 = { (int)0x00020000,      // workgroup_mask=0, data_size=4B
                 (int)(128u << 16),    // tensor_dim0 = 128
                 (int)(16u  << 16),    // tensor_dim1 = 16
                 (int)(128u << 16),    // tile_dim0 = 128
                 16,                   // tile_dim1 = 16
                 (int)(G_ * HD_),      // tensor_dim0_stride = 1024 elems
                 0, 0 };
      v4i gz4 = { 0, 0, 0, 0 };
      v8i gz8 = { 0, 0, 0, 0, 0, 0, 0, 0 };
      // 6-arg toolchain variant: (g0, g1, g2, g3, g4, cpol)
      __builtin_amdgcn_tensor_load_to_lds(g0, g1, gz4, gz4, gz8, 0);
    }
#endif

    // --- scores via WMMA (two independent chains) ---
    const int phys = blk * 16 + lo;                 // physical slot of this lane's key
    const float* krow = (phys == slot) ? knr : kc + ((size_t)phys * G_ + g) * HD_;
    v16bf bk[4];
    #pragma unroll
    for (int cch = 0; cch < 4; ++cch)
      cvt16(bk[cch], krow + cch * 32 + hi * 16);
    v8f sA = {}, sB = {};
    sA = __builtin_amdgcn_wmma_f32_16x16x32_bf16(false, aq[0], false, bk[0], (short)0, sA, false, false);
    sB = __builtin_amdgcn_wmma_f32_16x16x32_bf16(false, aq[1], false, bk[1], (short)0, sB, false, false);
    sA = __builtin_amdgcn_wmma_f32_16x16x32_bf16(false, aq[2], false, bk[2], (short)0, sA, false, false);
    sB = __builtin_amdgcn_wmma_f32_16x16x32_bf16(false, aq[3], false, bk[3], (short)0, sB, false, false);
    v8f sc = sA + sB;

    // --- online softmax (real scores: lanes 0-15, VGPR h = head h) ---
    const int  s     = s0 + lo;
    const bool valid = (s < ctx);
    float pv[RHEADS];
    #pragma unroll
    for (int h = 0; h < RHEADS; ++h) {
      float v = (hi == 0 && valid) ? sc[h] : -1e30f;
      v = fmaxf(v, __shfl_xor(v, 1));
      v = fmaxf(v, __shfl_xor(v, 2));
      v = fmaxf(v, __shfl_xor(v, 4));
      v = fmaxf(v, __shfl_xor(v, 8));
      v = fmaxf(v, __shfl_xor(v, 16));              // all lanes: tile max over valid keys
      const float m_new   = fmaxf(m_run[h], v);
      const float rescale = __expf(m_run[h] - m_new);
      m_run[h] = m_new;
      float p = (hi == 0 && valid) ? __expf(sc[h] - m_new) : 0.f;
      pv[h] = p;
      float srt = p;
      srt += __shfl_xor(srt, 1);
      srt += __shfl_xor(srt, 2);
      srt += __shfl_xor(srt, 4);
      srt += __shfl_xor(srt, 8);
      srt += __shfl_xor(srt, 16);                   // row sum (hi lanes contributed 0)
      l_run[h] = l_run[h] * rescale + srt;
      acc[h][0] *= rescale; acc[h][1] *= rescale;
      acc[h][2] *= rescale; acc[h][3] *= rescale;
    }
    if (hi == 0) {
      float4 p4; p4.x = pv[0]; p4.y = pv[1]; p4.z = pv[2]; p4.w = pv[3];
      lds_p[w][lo] = p4;                            // ds_store_b128
    }

#ifdef USE_TDM
    __builtin_amdgcn_s_wait_tensorcnt(0);           // V tile resident in LDS
    // Patch the new-token row if it lives in this tile.
    const int rel = slot - blk * 16;
    if (rel >= 0 && rel < 16)
      *(float4*)&lds_v[w][rel][lane * 4] = *(const float4*)(vnr + lane * 4);
#endif
    asm volatile("s_wait_dscnt 0" ::: "memory");    // same-wave LDS handoff

    // --- AV: each lane owns dims [lane*4, lane*4+4) for all 4 heads ---
    #pragma unroll 4
    for (int kk = 0; kk < 16; ++kk) {
      float4 p4 = lds_p[w][kk];                     // broadcast read
#ifdef USE_TDM
      float4 vv = *(const float4*)&lds_v[w][kk][lane * 4];
#else
      const int pk = blk * 16 + kk;
      const float* vrow = (pk == slot) ? vnr : vc + ((size_t)pk * G_ + g) * HD_;
      float4 vv = *(const float4*)(vrow + lane * 4);
#endif
      acc[0][0] += p4.x * vv.x; acc[0][1] += p4.x * vv.y;
      acc[0][2] += p4.x * vv.z; acc[0][3] += p4.x * vv.w;
      acc[1][0] += p4.y * vv.x; acc[1][1] += p4.y * vv.y;
      acc[1][2] += p4.y * vv.z; acc[1][3] += p4.y * vv.w;
      acc[2][0] += p4.z * vv.x; acc[2][1] += p4.z * vv.y;
      acc[2][2] += p4.z * vv.z; acc[2][3] += p4.z * vv.w;
      acc[3][0] += p4.w * vv.x; acc[3][1] += p4.w * vv.y;
      acc[3][2] += p4.w * vv.z; acc[3][3] += p4.w * vv.w;
    }
  }

  // --- publish per-wave partials ---
  #pragma unroll
  for (int h = 0; h < RHEADS; ++h) {
    lds_acc[w][h][lane * 4 + 0] = acc[h][0];
    lds_acc[w][h][lane * 4 + 1] = acc[h][1];
    lds_acc[w][h][lane * 4 + 2] = acc[h][2];
    lds_acc[w][h][lane * 4 + 3] = acc[h][3];
  }
  if (lane == 0) {
    #pragma unroll
    for (int h = 0; h < RHEADS; ++h) { lds_m[w][h] = m_run[h]; lds_l[w][h] = l_run[h]; }
  }
  __syncthreads();

  // --- flash-decoding combine: thread -> (head = tid>>5, dims = (tid&31)*4) ---
  const int h  = threadIdx.x >> 5;
  const int dl = (threadIdx.x & 31) * 4;
  float M = fmaxf(fmaxf(lds_m[0][h], lds_m[1][h]), fmaxf(lds_m[2][h], lds_m[3][h]));
  float L = 0.f, o0 = 0.f, o1 = 0.f, o2 = 0.f, o3 = 0.f;
  #pragma unroll
  for (int ww = 0; ww < 4; ++ww) {
    float f = __expf(lds_m[ww][h] - M);
    L  += f * lds_l[ww][h];
    o0 += f * lds_acc[ww][h][dl + 0];
    o1 += f * lds_acc[ww][h][dl + 1];
    o2 += f * lds_acc[ww][h][dl + 2];
    o3 += f * lds_acc[ww][h][dl + 3];
  }
  const float inv = 1.0f / L;
  float* op = o + (size_t)(b * H_ + g * RHEADS + h) * HD_ + dl;
  op[0] = o0 * inv; op[1] = o1 * inv; op[2] = o2 * inv; op[3] = o3 * inv;
}

// ---------------------------------------------------------------------------
extern "C" void kernel_launch(void* const* d_in, const int* in_sizes, int n_in,
                              void* d_out, int out_size, void* d_ws, size_t ws_size,
                              hipStream_t stream)
{
  (void)in_sizes; (void)n_in; (void)out_size; (void)ws_size;
  const float* x      = (const float*)d_in[0];
  const float* wq     = (const float*)d_in[1];
  const float* wk     = (const float*)d_in[2];
  const float* wv     = (const float*)d_in[3];
  const float* wo     = (const float*)d_in[4];
  const float* kcache = (const float*)d_in[5];
  const float* vcache = (const float*)d_in[6];
  const float* cosb   = (const float*)d_in[7];
  const float* sinb   = (const float*)d_in[8];
  const int*   slotmap = (const int*)d_in[9];
  const int*   ctxlen  = (const int*)d_in[10];
  const int*   btab    = (const int*)d_in[11];
  float* out = (float*)d_out;

  // Workspace: q (512KB) | k_new (128KB) | v_new (128KB) | o (512KB)
  float* q_ws = (float*)d_ws;
  float* k_ws = q_ws + (size_t)B_ * H_ * HD_;
  float* v_ws = k_ws + (size_t)B_ * G_ * HD_;
  float* o_ws = v_ws + (size_t)B_ * G_ * HD_;

  // QKV projections (WMMA bf16, fp32 accumulate)
  gemm_xwT_bf16<<<64, 256, 0, stream>>>(x, wq, q_ws, H_ * HD_, NIN);  // 512 tiles
  gemm_xwT_bf16<<<16, 256, 0, stream>>>(x, wk, k_ws, G_ * HD_, NIN);  // 128 tiles
  gemm_xwT_bf16<<<16, 256, 0, stream>>>(x, wv, v_ws, G_ * HD_, NIN);

  // RoPE on q (+score scale) and k_new
  rope_scale<<<dim3(B_, H_ + G_), 64, 0, stream>>>(q_ws, k_ws, cosb, sinb);

  // Paged attention (read-only caches; substitutes the new token's row)
  paged_attn<<<dim3(B_, G_), 128, 0, stream>>>(q_ws, k_ws, v_ws, kcache, vcache,
                                               btab, ctxlen, slotmap, o_ws);

  // Output projection
  gemm_xwT_bf16<<<64, 256, 0, stream>>>(o_ws, wo, out, NIN, H_ * HD_);
}
